// PAdicLinearLayer_90460601188760
// MI455X (gfx1250) — compile-verified, matrix operand order checked by reading, not run on hardware
//
#include <hip/hip_runtime.h>

typedef __attribute__((ext_vector_type(4)))  __bf16 v4bf;
typedef __attribute__((ext_vector_type(8)))  __bf16 v8bf;
typedef __attribute__((ext_vector_type(16))) __bf16 v16bf;
typedef __attribute__((ext_vector_type(8)))  float  v8f;
typedef __attribute__((ext_vector_type(4)))  int    v4i;

#define AS1 __attribute__((address_space(1)))
#define AS3 __attribute__((address_space(3)))

#if defined(__has_builtin)
#if __has_builtin(__builtin_amdgcn_global_load_async_to_lds_b128) && \
    __has_builtin(__builtin_amdgcn_s_wait_asynccnt)
#define USE_ASYNC_LDS 1
#endif
#endif
#ifndef USE_ASYNC_LDS
#define USE_ASYNC_LDS 0
#endif

#define B_SZ   32768
#define DI     512
#define DO     512
#define NL     10
#define TILE_M 128
#define TILE_N 128
#define TILE_K 64
#define LDSK   72                     // 64 + 8 bf16 pad (keeps 16B alignment)
#define MAX_TILES 272
#define PERM_CAP (B_SZ + NL * TILE_M) // 34048

// workspace layout (byte offsets)
#define COEF_OFF 0                    // 100 floats
#define META_OFF 1024                 // ints
#define PERM_OFF 8192                 // PERM_CAP ints
#define WEFF_OFF 147456               // NL*DO*DI bf16  (~5.25 MB total ws use)

// meta[] int indices
#define M_COUNT  0
#define M_CURSOR 10
#define M_NTILES 20
#define M_TCLASS 32
#define M_TBASE  (32 + MAX_TILES)

__device__ __forceinline__ int clamp_lvl(int v) {
    return v < 0 ? 0 : (v > NL - 1 ? NL - 1 : v);
}

// ---- 1. per-class combine coefficients: c[v][l] -------------------------
__global__ void coef_kernel(const float* __restrict__ gates,
                            const float* __restrict__ attn,
                            float* __restrict__ coef) {
    int t = threadIdx.x;
    if (t >= NL * NL) return;
    int v = t / NL, l = t % NL;
    float m = attn[v * NL];
    for (int j = 1; j < NL; ++j) m = fmaxf(m, attn[v * NL + j]);
    float s = 0.f;
    for (int j = 0; j < NL; ++j) s += __expf(attn[v * NL + j] - m);
    float a  = __expf(attn[v * NL + l] - m) / s;
    float gl = 1.f / (1.f + __expf(-gates[l]));
    float c  = 0.3f * a * gl;
    if (l == v) c += 0.7f * gl;
    coef[t] = c;
}

// ---- 2. Weff[v] = sum_l c[v][l] * W[l], stored bf16 ---------------------
__global__ __launch_bounds__(256) void weff_kernel(const float* __restrict__ W,
                                                   const float* __restrict__ coef,
                                                   __bf16* __restrict__ weff) {
    __shared__ float sc[NL * NL];
    int tid = threadIdx.x;
    if (tid < NL * NL) sc[tid] = coef[tid];
    __syncthreads();
    int e = blockIdx.x * 256 + tid;   // 0 .. DO*DI-1
    float w[NL];
#pragma unroll
    for (int l = 0; l < NL; ++l) w[l] = W[(size_t)l * DO * DI + e];
#pragma unroll
    for (int v = 0; v < NL; ++v) {
        float acc = 0.f;
#pragma unroll
        for (int l = 0; l < NL; ++l) acc += sc[v * NL + l] * w[l];
        weff[(size_t)v * DO * DI + e] = (__bf16)acc;
    }
}

// ---- 3. counting sort of samples by class -------------------------------
__global__ void init_kernel(int* __restrict__ meta, int* __restrict__ perm) {
    int tid = blockIdx.x * blockDim.x + threadIdx.x;
    if (tid < PERM_CAP) perm[tid] = -1;
    if (tid < NL) meta[M_COUNT + tid] = 0;
}

__global__ void hist_kernel(const int* __restrict__ val, int* __restrict__ meta) {
    int b = blockIdx.x * blockDim.x + threadIdx.x;
    if (b >= B_SZ) return;
    atomicAdd(&meta[M_COUNT + clamp_lvl(val[b])], 1);
}

__global__ void scan_kernel(int* __restrict__ meta) {
    if (threadIdx.x != 0 || blockIdx.x != 0) return;
    int row = 0, nt = 0;
    for (int v = 0; v < NL; ++v) {
        int c = meta[M_COUNT + v];
        meta[M_CURSOR + v] = row;                       // segment start / cursor
        int t = (c + TILE_M - 1) / TILE_M;
        for (int i = 0; i < t; ++i) {
            meta[M_TCLASS + nt] = v;
            meta[M_TBASE + nt]  = row + i * TILE_M;
            ++nt;
        }
        row += t * TILE_M;
    }
    meta[M_NTILES] = nt;
}

__global__ void scatter_kernel(const int* __restrict__ val,
                               int* __restrict__ meta, int* __restrict__ perm) {
    int b = blockIdx.x * blockDim.x + threadIdx.x;
    if (b >= B_SZ) return;
    int v = clamp_lvl(val[b]);
    int pos = atomicAdd(&meta[M_CURSOR + v], 1);
    perm[pos] = b;
}

// ---- 4. grouped GEMM: out[perm] = bf16(x[perm]) * Weff[class]^T + bias --
__global__ __launch_bounds__(256) void gemm_kernel(const float* __restrict__ x,
                                                   const __bf16* __restrict__ weff,
                                                   const float* __restrict__ bias,
                                                   const int* __restrict__ meta,
                                                   const int* __restrict__ perm,
                                                   float* __restrict__ out) {
    __shared__ __bf16 sA[TILE_M * LDSK];
    __shared__ __bf16 sB[TILE_N * LDSK];
    __shared__ int    sPerm[TILE_M];

    int tileId = blockIdx.y;
    if (tileId >= meta[M_NTILES]) return;
    int vcls    = meta[M_TCLASS + tileId];
    int rowbase = meta[M_TBASE + tileId];

    int tid  = threadIdx.x;
    int lane = tid & 31;          // wave32
    int wid  = tid >> 5;          // 8 waves
    int wm   = wid & 1;           // 64-row half of the M tile
    int wn   = wid >> 1;          // 32-col quarter of the N tile
    int lr   = lane & 15;
    int kb   = (lane < 16) ? 0 : 8;   // 16-bit A/B fragment K-base per lane half

    if (tid < TILE_M) sPerm[tid] = perm[rowbase + tid];

    const __bf16* wv = weff + (size_t)vcls * DO * DI;
    int o0 = blockIdx.x * TILE_N;

    v8f zero = {0.f, 0.f, 0.f, 0.f, 0.f, 0.f, 0.f, 0.f};
    v8f acc[4][2];
#pragma unroll
    for (int mt = 0; mt < 4; ++mt)
#pragma unroll
        for (int nt = 0; nt < 2; ++nt) acc[mt][nt] = zero;

    for (int k0 = 0; k0 < DI; k0 += TILE_K) {
        __syncthreads();   // previous iteration finished reading LDS (also covers sPerm)
        // stage B: 128 output rows x 64 k of Weff (already bf16)
#if USE_ASYNC_LDS
#pragma unroll
        for (int it = 0; it < 4; ++it) {
            int chunk = tid + it * 256;      // 0..1023 8-elem chunks
            int r  = chunk >> 3;
            int c8 = (chunk & 7) << 3;
            AS1 v4i* gsrc =
                (AS1 v4i*)(unsigned long long)(wv + (size_t)(o0 + r) * DI + k0 + c8);
            AS3 v4i* ldst =
                (AS3 v4i*)(unsigned)(unsigned long long)&sB[r * LDSK + c8];
            __builtin_amdgcn_global_load_async_to_lds_b128(gsrc, ldst, 0, 0);
        }
#else
#pragma unroll
        for (int it = 0; it < 4; ++it) {
            int chunk = tid + it * 256;      // 0..1023 8-elem chunks
            int r  = chunk >> 3;
            int c8 = (chunk & 7) << 3;
            *(v8bf*)&sB[r * LDSK + c8] =
                *(const v8bf*)(wv + (size_t)(o0 + r) * DI + k0 + c8);
        }
#endif
        // stage A: gather 128 rows x 64 k of x, fp32 -> bf16
#pragma unroll
        for (int it = 0; it < 8; ++it) {
            int vec = tid + it * 256;        // 0..2047 float4 chunks
            int r   = vec >> 4;
            int c4  = (vec & 15) << 2;
            int g   = sPerm[r];
            float4 f = make_float4(0.f, 0.f, 0.f, 0.f);
            if (g >= 0) f = *(const float4*)(x + (size_t)g * DI + k0 + c4);
            v4bf t = {(__bf16)f.x, (__bf16)f.y, (__bf16)f.z, (__bf16)f.w};
            *(v4bf*)&sA[r * LDSK + c4] = t;
        }
#if USE_ASYNC_LDS
        __builtin_amdgcn_s_wait_asynccnt(0);
#endif
        __syncthreads();

#pragma unroll
        for (int kk = 0; kk < TILE_K; kk += 32) {
            v16bf afr[4], bfr[2];
#pragma unroll
            for (int mt = 0; mt < 4; ++mt) {
                int row = wm * 64 + mt * 16 + lr;
                v8bf lo = *(const v8bf*)&sA[row * LDSK + kk + kb];
                v8bf hi = *(const v8bf*)&sA[row * LDSK + kk + kb + 16];
                afr[mt] = __builtin_shufflevector(lo, hi, 0, 1, 2, 3, 4, 5, 6, 7,
                                                  8, 9, 10, 11, 12, 13, 14, 15);
            }
#pragma unroll
            for (int nt = 0; nt < 2; ++nt) {
                int col = wn * 32 + nt * 16 + lr;
                v8bf lo = *(const v8bf*)&sB[col * LDSK + kk + kb];
                v8bf hi = *(const v8bf*)&sB[col * LDSK + kk + kb + 16];
                bfr[nt] = __builtin_shufflevector(lo, hi, 0, 1, 2, 3, 4, 5, 6, 7,
                                                  8, 9, 10, 11, 12, 13, 14, 15);
            }
#pragma unroll
            for (int mt = 0; mt < 4; ++mt)
#pragma unroll
                for (int nt = 0; nt < 2; ++nt)
                    acc[mt][nt] = __builtin_amdgcn_wmma_f32_16x16x32_bf16(
                        false, afr[mt], false, bfr[nt],
                        (short)0, acc[mt][nt], false, false);
        }
    }

    // epilogue: +bias, scatter-store valid rows
    int mhalf = (lane < 16) ? 0 : 8;
#pragma unroll
    for (int nt = 0; nt < 2; ++nt) {
        int col = o0 + wn * 32 + nt * 16 + lr;
        float bv = bias[col];
#pragma unroll
        for (int mt = 0; mt < 4; ++mt) {
#pragma unroll
            for (int r = 0; r < 8; ++r) {
                int rowl = wm * 64 + mt * 16 + r + mhalf;
                int g = sPerm[rowl];
                if (g >= 0) out[(size_t)g * DO + col] = acc[mt][nt][r] + bv;
            }
        }
    }
}

extern "C" void kernel_launch(void* const* d_in, const int* in_sizes, int n_in,
                              void* d_out, int out_size, void* d_ws, size_t ws_size,
                              hipStream_t stream) {
    const float* x     = (const float*)d_in[0];
    const int*   val   = (const int*)d_in[1];
    const float* W     = (const float*)d_in[2];
    const float* gates = (const float*)d_in[3];
    const float* attn  = (const float*)d_in[4];
    const float* bias  = (const float*)d_in[5];
    float* out = (float*)d_out;

    char* ws = (char*)d_ws;
    float*  coef = (float*)(ws + COEF_OFF);
    int*    meta = (int*)(ws + META_OFF);
    int*    perm = (int*)(ws + PERM_OFF);
    __bf16* weff = (__bf16*)(ws + WEFF_OFF);

    coef_kernel<<<1, 128, 0, stream>>>(gates, attn, coef);
    weff_kernel<<<(DO * DI) / 256, 256, 0, stream>>>(W, coef, weff);
    init_kernel<<<(PERM_CAP + 255) / 256, 256, 0, stream>>>(meta, perm);
    hist_kernel<<<B_SZ / 256, 256, 0, stream>>>(val, meta);
    scan_kernel<<<1, 1, 0, stream>>>(meta);
    scatter_kernel<<<B_SZ / 256, 256, 0, stream>>>(val, meta, perm);
    gemm_kernel<<<dim3(DO / TILE_N, MAX_TILES), 256, 0, stream>>>(
        x, weff, bias, meta, perm, out);
}
